// ExpertLSTM_47132971106933
// MI455X (gfx1250) — matrix operand order repaired and checked by reading, not compiled, not run on hardware
//
#include <hip/hip_runtime.h>
#include <hip/hip_bf16.h>

// Problem constants (match reference)
#define BATCH 8192
#define SEQ   128
#define INSZ  32
#define HID   64
#define G4    256   // 4*HID gate width

// LDS row strides (elements), padded to break bank conflicts, keep 16B align
#define KP_IH0 40   // 32 + 8
#define KP_H   72   // 64 + 8

typedef __attribute__((ext_vector_type(16))) __bf16 v16bf;
typedef __attribute__((ext_vector_type(8)))  float  v8f;

// gfx1250 has a hardware v_tanh_f32 transcendental.
__device__ __forceinline__ float fast_tanh(float x) {
#if __has_builtin(__builtin_amdgcn_tanhf)
  return __builtin_amdgcn_tanhf(x);
#else
  return tanhf(x);
#endif
}
// sigmoid(x) == 0.5 + 0.5*tanh(0.5*x): mul + tanh + fma
__device__ __forceinline__ float sigf(float x) {
  return fmaf(0.5f, fast_tanh(0.5f * x), 0.5f);
}

// ---- A fragment (16x32 bf16) from LDS, row-major source, stride in elements.
// Lane l: row m=l&15; lanes 0-15 hold K {k0+0..7, k0+16..23}, lanes 16-31 hold
// K {k0+8..15, k0+24..31} (ISA 7.12.2, 16-bit A 16x32).
__device__ __forceinline__ v16bf lds_a_frag(const __bf16* base, int stride, int lane, int k0) {
  const int m  = lane & 15;
  const int hi = lane >> 4;
  const __bf16* p = base + m * stride + k0 + hi * 8;
  v16bf a;
  ((uint4*)&a)[0] = *(const uint4*)(p);        // 8 bf16: K +0..7 (or +8..15)
  ((uint4*)&a)[1] = *(const uint4*)(p + 16);   // 8 bf16: K +16..23 (or +24..31)
  return a;
}

// ---- B fragment (32x16 bf16) from LDS. Source is W[n][k] row-major; WMMA B is
// B[k][n] = W[n0+n][k]. Lane l: column n=n0+(l&15); lanes 0-15 hold K {k0+0..7, k0+16..23},
// lanes 16-31 hold K {k0+8..15, k0+24..31} (matches A fragment).
__device__ __forceinline__ v16bf lds_b_frag(const __bf16* w, int stride, int lane, int n0, int k0) {
  const int n  = n0 + (lane & 15);
  const int kk = k0 + ((lane >> 4) << 3);
  const __bf16* p = w + n * stride + kk;
  v16bf b;
  ((uint4*)&b)[0] = *(const uint4*)(p);
  ((uint4*)&b)[1] = *(const uint4*)(p + 16);
  return b;
}

// Fully fused 2-layer LSTM + linear head. One wave owns 16 batch rows for the
// whole sequence; weights live in LDS (bf16), c-state in fp32 VGPRs, h-state in
// per-wave in-place LDS buffers (bf16). In-place is safe: each step loads ALL
// h A-fragments before issuing any h stores, and same-wave LDS ops are
// architecturally in-order (DScnt).
// LDS total = 151,552 B per WG -> two workgroups fit in a 320 KB WGP.
__global__ void __launch_bounds__(128)
__attribute__((amdgpu_waves_per_eu(4)))   // pin VGPR budget <= 256: no vgpr-msb traffic
lstm2_wmma_fused(const float* __restrict__ x,
                 const float* __restrict__ w_ih0, const float* __restrict__ w_hh0,
                 const float* __restrict__ b_ih0, const float* __restrict__ b_hh0,
                 const float* __restrict__ w_ih1, const float* __restrict__ w_hh1,
                 const float* __restrict__ b_ih1, const float* __restrict__ b_hh1,
                 const float* __restrict__ fc_w,  const float* __restrict__ fc_b,
                 float* __restrict__ out) {
  extern __shared__ __align__(16) char smem[];
  __bf16* l_wih0 = (__bf16*)smem;                    // 256*40 el
  __bf16* l_whh0 = l_wih0 + G4 * KP_IH0;             // 256*72 el
  __bf16* l_wih1 = l_whh0 + G4 * KP_H;               // 256*72 el
  __bf16* l_whh1 = l_wih1 + G4 * KP_H;               // 256*72 el
  float*  l_b0   = (float*)(l_whh1 + G4 * KP_H);     // 256 f32 (b_ih0+b_hh0)
  float*  l_b1   = l_b0 + G4;                        // 256 f32
  __bf16* l_h    = (__bf16*)(l_b1 + G4);             // per-wave h state

  const int tid  = threadIdx.x;
  const int lane = tid & 31;
  const int wave = tid >> 5;
  const int nthr = blockDim.x;

  // ---- Stage weights (fp32 -> bf16) and combined biases into LDS ----
  for (int i = tid; i < G4 * INSZ; i += nthr)
    l_wih0[(i / INSZ) * KP_IH0 + (i % INSZ)] = (__bf16)w_ih0[i];
  for (int i = tid; i < G4 * HID; i += nthr)
    l_whh0[(i / HID) * KP_H + (i % HID)] = (__bf16)w_hh0[i];
  for (int i = tid; i < G4 * HID; i += nthr)
    l_wih1[(i / HID) * KP_H + (i % HID)] = (__bf16)w_ih1[i];
  for (int i = tid; i < G4 * HID; i += nthr)
    l_whh1[(i / HID) * KP_H + (i % HID)] = (__bf16)w_hh1[i];
  for (int i = tid; i < G4; i += nthr) {
    l_b0[i] = b_ih0[i] + b_hh0[i];
    l_b1[i] = b_ih1[i] + b_hh1[i];
  }
  __syncthreads();  // only block-wide sync; time loop is wave-private

  // Per-wave h state: [layer][16 rows][KP_H], bf16, updated in place
  __bf16* h0buf = l_h + wave * (2 * 16 * KP_H);
  __bf16* h1buf = h0buf + 16 * KP_H;
  for (int i = lane; i < 16 * KP_H; i += 32) {   // zero t=0 state
    h0buf[i] = (__bf16)0.0f;
    h1buf[i] = (__bf16)0.0f;
  }

  v8f c0[4], c1[4];  // fp32 cell state, 4 column-quads of 16
  v8f vzero;         // zero C-operand (folds to WMMA inline-0 SRC2)
  #pragma unroll
  for (int r = 0; r < 8; ++r) vzero[r] = 0.0f;
  #pragma unroll
  for (int q = 0; q < 4; ++q) {
    c0[q] = vzero;
    c1[q] = vzero;
  }

  const int wrow0 = blockIdx.x * ((nthr >> 5) * 16) + wave * 16;  // batch base
  const int mrow  = wrow0 + (lane & 15);   // this lane's A-matrix row
  const int mloc  = (lane >> 4) * 8;       // C/D: VGPR r holds row mloc+r
  const int ncol  = lane & 15;             // C/D: this lane's column in tile
  const float* xlane = x + (size_t)mrow * SEQ * INSZ + (lane >> 4) * 8;

  for (int t = 0; t < SEQ; ++t) {
    // Prefetch next timestep's x row (global_prefetch_b8)
    if (t + 1 < SEQ) __builtin_prefetch(xlane + (size_t)(t + 1) * INSZ, 0, 1);

    // ======== Layer 0 ========
    // x_t A-fragment straight from global (fp32 -> bf16), vector loads
    v16bf a_x;
    {
      const float4* pv = (const float4*)(xlane + (size_t)t * INSZ);
      const float4 q0 = pv[0], q1 = pv[1];   // K +0..7   (or +8..15)
      const float4 q2 = pv[4], q3 = pv[5];   // K +16..23 (or +24..31)
      __bf16* tp = (__bf16*)&a_x;
      tp[0]  = (__bf16)q0.x; tp[1]  = (__bf16)q0.y; tp[2]  = (__bf16)q0.z; tp[3]  = (__bf16)q0.w;
      tp[4]  = (__bf16)q1.x; tp[5]  = (__bf16)q1.y; tp[6]  = (__bf16)q1.z; tp[7]  = (__bf16)q1.w;
      tp[8]  = (__bf16)q2.x; tp[9]  = (__bf16)q2.y; tp[10] = (__bf16)q2.z; tp[11] = (__bf16)q2.w;
      tp[12] = (__bf16)q3.x; tp[13] = (__bf16)q3.y; tp[14] = (__bf16)q3.z; tp[15] = (__bf16)q3.w;
    }
    // Load h0(t) fragments BEFORE any store of h0(t+1): in-place safe (in-order LDS)
    const v16bf a_h0a = lds_a_frag(h0buf, KP_H, lane, 0);
    const v16bf a_h0b = lds_a_frag(h0buf, KP_H, lane, 32);

    #pragma unroll
    for (int nq = 0; nq < 4; ++nq) {
      v8f acc[4];
      #pragma unroll
      for (int g = 0; g < 4; ++g) {                  // i, f, g, o
        const int n0 = g * HID + nq * 16;            // gate column tile base
        acc[g] = __builtin_amdgcn_wmma_f32_16x16x32_bf16(
            false, a_x,   false, lds_b_frag(l_wih0, KP_IH0, lane, n0, 0),  (short)0, vzero,  false, false);
        acc[g] = __builtin_amdgcn_wmma_f32_16x16x32_bf16(
            false, a_h0a, false, lds_b_frag(l_whh0, KP_H,   lane, n0, 0),  (short)0, acc[g], false, false);
        acc[g] = __builtin_amdgcn_wmma_f32_16x16x32_bf16(
            false, a_h0b, false, lds_b_frag(l_whh0, KP_H,   lane, n0, 32), (short)0, acc[g], false, false);
      }
      // biases folded into the activation epilogue (scalar adds, no C-splat)
      const float bi = l_b0[nq * 16 + ncol];
      const float bf_ = l_b0[HID + nq * 16 + ncol];
      const float bg = l_b0[2 * HID + nq * 16 + ncol];
      const float bo = l_b0[3 * HID + nq * 16 + ncol];
      #pragma unroll
      for (int r = 0; r < 8; ++r) {
        const float iv = sigf(acc[0][r] + bi);
        const float fv = sigf(acc[1][r] + bf_);
        const float gv = fast_tanh(acc[2][r] + bg);
        const float ov = sigf(acc[3][r] + bo);
        const float cv = fv * c0[nq][r] + iv * gv;
        c0[nq][r] = cv;
        h0buf[(mloc + r) * KP_H + nq * 16 + ncol] = (__bf16)(ov * fast_tanh(cv));
      }
    }

    // ======== Layer 1 (consumes fresh h0(t+1), recurrent on h1(t)) ========
    const v16bf a_i0  = lds_a_frag(h0buf, KP_H, lane, 0);
    const v16bf a_i1  = lds_a_frag(h0buf, KP_H, lane, 32);
    const v16bf a_h1a = lds_a_frag(h1buf, KP_H, lane, 0);
    const v16bf a_h1b = lds_a_frag(h1buf, KP_H, lane, 32);

    #pragma unroll
    for (int nq = 0; nq < 4; ++nq) {
      v8f acc[4];
      #pragma unroll
      for (int g = 0; g < 4; ++g) {
        const int n0 = g * HID + nq * 16;
        acc[g] = __builtin_amdgcn_wmma_f32_16x16x32_bf16(
            false, a_i0,  false, lds_b_frag(l_wih1, KP_H, lane, n0, 0),  (short)0, vzero,  false, false);
        acc[g] = __builtin_amdgcn_wmma_f32_16x16x32_bf16(
            false, a_i1,  false, lds_b_frag(l_wih1, KP_H, lane, n0, 32), (short)0, acc[g], false, false);
        acc[g] = __builtin_amdgcn_wmma_f32_16x16x32_bf16(
            false, a_h1a, false, lds_b_frag(l_whh1, KP_H, lane, n0, 0),  (short)0, acc[g], false, false);
        acc[g] = __builtin_amdgcn_wmma_f32_16x16x32_bf16(
            false, a_h1b, false, lds_b_frag(l_whh1, KP_H, lane, n0, 32), (short)0, acc[g], false, false);
      }
      const float bi = l_b1[nq * 16 + ncol];
      const float bf_ = l_b1[HID + nq * 16 + ncol];
      const float bg = l_b1[2 * HID + nq * 16 + ncol];
      const float bo = l_b1[3 * HID + nq * 16 + ncol];
      #pragma unroll
      for (int r = 0; r < 8; ++r) {
        const float iv = sigf(acc[0][r] + bi);
        const float fv = sigf(acc[1][r] + bf_);
        const float gv = fast_tanh(acc[2][r] + bg);
        const float ov = sigf(acc[3][r] + bo);
        const float cv = fv * c1[nq][r] + iv * gv;
        c1[nq][r] = cv;
        h1buf[(mloc + r) * KP_H + nq * 16 + ncol] = (__bf16)(ov * fast_tanh(cv));
      }
    }
  }

  // ---- Head: logit[m] = fc_b + sum_c h1_final[m][c] * fc_w[c] ----
  if (lane < 16) {
    const __bf16* hf = h1buf + lane * KP_H;
    float sum = fc_b[0];
    #pragma unroll
    for (int c = 0; c < HID; ++c) sum += (float)hf[c] * fc_w[c];
    out[wrow0 + lane] = sum;
  }
}

extern "C" void kernel_launch(void* const* d_in, const int* in_sizes, int n_in,
                              void* d_out, int out_size, void* d_ws, size_t ws_size,
                              hipStream_t stream) {
  const float* x     = (const float*)d_in[0];
  const float* w_ih0 = (const float*)d_in[1];
  const float* w_hh0 = (const float*)d_in[2];
  const float* b_ih0 = (const float*)d_in[3];
  const float* b_hh0 = (const float*)d_in[4];
  const float* w_ih1 = (const float*)d_in[5];
  const float* w_hh1 = (const float*)d_in[6];
  const float* b_ih1 = (const float*)d_in[7];
  const float* b_hh1 = (const float*)d_in[8];
  const float* fc_w  = (const float*)d_in[9];
  const float* fc_b  = (const float*)d_in[10];
  float* out = (float*)d_out;

  const int block = 128;                       // 4 waves -> 64 batch rows / WG
  const int grid  = BATCH / 64;                // 128 workgroups
  // Dynamic LDS: weights(bf16,padded) + biases(f32) + per-wave in-place h state
  // = 131072 + 2048 + 4*4608 = 151,552 B  (two WGs per 320 KB WGP)
  const size_t lds = (size_t)(G4 * KP_IH0 + 3 * G4 * KP_H) * sizeof(__bf16)
                   + (size_t)(2 * G4) * sizeof(float)
                   + (size_t)(block / 32) * (2 * 16 * KP_H) * sizeof(__bf16);

  lstm2_wmma_fused<<<grid, block, lds, stream>>>(
      x, w_ih0, w_hh0, b_ih0, b_hh0, w_ih1, w_hh1, b_ih1, b_hh1, fc_w, fc_b, out);
}
